// MultiHeadAttention_51762945851506
// MI455X (gfx1250) — matrix operand order repaired and checked by reading, not compiled, not run on hardware
//
#include <hip/hip_runtime.h>
#include <hip/hip_bf16.h>

// ---------------------------------------------------------------------------
// MHA forward for MI455X (gfx1250, wave32, WMMA bf16, TDM async staging).
//   B=4, S=2048, E=1024, H=16, HD=64
// Pipeline: f32->bf16 convert -> QKV GEMM (TDM-staged B panel, scatter to
// Q/K/Vt) -> flash attention (online softmax) -> projection GEMM (+bias, f32).
// ---------------------------------------------------------------------------

#define BATCH 4
#define SEQ   2048
#define EMB   1024
#define NH    16
#define HD    64
#define NQKV  (3 * EMB)       // 3072
#define ROWS  (BATCH * SEQ)   // 8192

typedef __attribute__((ext_vector_type(16))) __bf16 v16bf;
typedef __attribute__((ext_vector_type(8)))  __bf16 bf16x8;
typedef __attribute__((ext_vector_type(8)))  float  v8f;
typedef __attribute__((ext_vector_type(4)))  unsigned int v4u32;
typedef __attribute__((ext_vector_type(4)))  int v4i32;
typedef __attribute__((ext_vector_type(8)))  int v8i32;

#ifndef __has_builtin
#define __has_builtin(x) 0
#endif
#if defined(__HIP_DEVICE_COMPILE__) && \
    __has_builtin(__builtin_amdgcn_tensor_load_to_lds)
#define HAVE_TDM 1
#else
#define HAVE_TDM 0
#endif

static __device__ __forceinline__ v8f wmma_bf16(v16bf a, v16bf b, v8f c) {
  // D = A(16x32 bf16) * B(32x16 bf16) + C(16x16 f32)
  return __builtin_amdgcn_wmma_f32_16x16x32_bf16(
      /*neg_a=*/false, a, /*neg_b=*/false, b,
      /*c_mod=*/(short)0, c, /*reuse_a=*/false, /*reuse_b=*/false);
}

// A-fragment (16x32, K-contiguous rows): lane holds row = lane%16,
// 16 contiguous K values {k0..k0+15}, k0 = (lane<16 ? 0 : 16).
static __device__ __forceinline__ v16bf load_a_frag(const __bf16* p) {
  bf16x8 lo = *(const bf16x8*)(p);
  bf16x8 hi = *(const bf16x8*)(p + 8);
  return __builtin_shufflevector(lo, hi, 0, 1, 2, 3, 4, 5, 6, 7,
                                 8, 9, 10, 11, 12, 13, 14, 15);
}

// B-fragment (32x16): lane holds column = lane%16, 16 contiguous K values
// starting at (lane<16 ? 0 : 16).  `p` must be 32B aligned.
static __device__ __forceinline__ v16bf load_b_frag(const __bf16* p) {
  return *(const v16bf*)(p);
}

// ---------------------------------------------------------------------------
// TDM: async DMA of a 2D bf16 tile (tile_cols x tile_rows) from global into
// LDS (row-major, rows of tile_cols elements).  Tracked by TENSORcnt.
// D# layout per CDNA5 ISA.
// ---------------------------------------------------------------------------
static __device__ __forceinline__ void tdm_load_tile_bf16(
    unsigned lds_off, const __bf16* gsrc, unsigned tile_cols,
    unsigned tile_rows, unsigned tensor_cols, unsigned tensor_rows,
    unsigned row_stride_elems) {
#if HAVE_TDM
  const unsigned long long ga = (unsigned long long)gsrc;
  v4u32 g0;
  g0[0] = 1u;                                  // count=1 (valid), user mode
  g0[1] = lds_off;                             // lds_addr (bytes)
  g0[2] = (unsigned)ga;                        // global_addr[31:0]
  g0[3] = (unsigned)(ga >> 32) | (2u << 30);   // global_addr[56:32] | type=2
  v8i32 g1;
  g1[0] = (int)(1u << 16);                     // data_size = 2 bytes
  g1[1] = (int)((tensor_cols & 0xffffu) << 16);            // tensor_dim0 lo
  g1[2] = (int)((tensor_cols >> 16) |
                ((tensor_rows & 0xffffu) << 16));          // dim0 hi | dim1 lo
  g1[3] = (int)((tensor_rows >> 16) | (tile_cols << 16));  // dim1 hi | tile0
  g1[4] = (int)tile_rows;                      // tile_dim1 (tile_dim2=0)
  g1[5] = (int)row_stride_elems;               // tensor_dim0_stride[31:0]
  g1[6] = 0;
  g1[7] = 0;
  v4i32 z4 = {0, 0, 0, 0};
#if __clang_major__ >= 23
  v8i32 z8 = {0, 0, 0, 0, 0, 0, 0, 0};
  __builtin_amdgcn_tensor_load_to_lds(g0, g1, z4, z4, z8, 0);
#else
  __builtin_amdgcn_tensor_load_to_lds(g0, g1, z4, z4, 0);
#endif
#else
  (void)lds_off; (void)gsrc; (void)tile_cols; (void)tile_rows;
  (void)tensor_cols; (void)tensor_rows; (void)row_stride_elems;
#endif
}

// ---------------------------------------------------------------------------
// f32 -> bf16 conversion
// ---------------------------------------------------------------------------
__global__ void cvt_f32_bf16(const float* __restrict__ in,
                             __bf16* __restrict__ out, int n) {
  int i = blockIdx.x * blockDim.x + threadIdx.x;
  if (i < n) out[i] = (__bf16)in[i];
}

// ---------------------------------------------------------------------------
// Shared GEMM mainloop: C(128x128 per block) = A @ W^T, A/W K-contiguous bf16.
// B panel (128 rows x 32 K) staged in LDS by TDM, double buffered.
// acc indexed [mt][nt]; wave tile 32(M) x 64(N).
// ---------------------------------------------------------------------------
#define GEMM_MAINLOOP(Aptr, Wpanel, NTOT)                                      \
  __shared__ __align__(128) __bf16 btile[2][128 * 32];                         \
  const int tid  = threadIdx.x;                                                \
  const int lane = tid & 31;                                                   \
  const int wave = tid >> 5;                                                   \
  const int l16  = lane & 15;                                                  \
  const bool hi  = lane >= 16;                                                 \
  const int Mbase = blockIdx.x * 128 + (wave & 3) * 32;                        \
  const int wn    = wave >> 2;                                                 \
  const int Nbase = blockIdx.y * 128 + wn * 64;                                \
  const __bf16* arow[2];                                                       \
  _Pragma("unroll") for (int mt = 0; mt < 2; ++mt)                             \
      arow[mt] = (Aptr) + (size_t)(Mbase + mt * 16 + l16) * EMB;               \
  v8f acc[2][4];                                                               \
  _Pragma("unroll") for (int mt = 0; mt < 2; ++mt)                             \
      _Pragma("unroll") for (int nt = 0; nt < 4; ++nt)                         \
          acc[mt][nt] = (v8f){0.f, 0.f, 0.f, 0.f, 0.f, 0.f, 0.f, 0.f};         \
  const int ka = hi ? 16 : 0;                                                  \
  const int kb = hi ? 16 : 0;                                                  \
  const unsigned lds0 =                                                        \
      (unsigned)(unsigned long long)(void*)&btile[0][0];                       \
  const unsigned lds1 =                                                        \
      (unsigned)(unsigned long long)(void*)&btile[1][0];                       \
  int cur = 0;                                                                 \
  if (HAVE_TDM && wave == 0)                                                   \
    tdm_load_tile_bf16(lds0, (Wpanel), 32, 128, EMB, (NTOT), EMB);             \
  for (int k0 = 0; k0 < EMB; k0 += 32) {                                       \
    if (HAVE_TDM) {                                                            \
      if (wave == 0) {                                                         \
        const bool more = (k0 + 32) < EMB;                                     \
        if (more)                                                              \
          tdm_load_tile_bf16(cur ? lds0 : lds1, (Wpanel) + k0 + 32, 32, 128,   \
                             EMB, (NTOT), EMB);                                \
        if (more)                                                              \
          __builtin_amdgcn_s_wait_tensorcnt((short)1);                         \
        else                                                                   \
          __builtin_amdgcn_s_wait_tensorcnt((short)0);                         \
      }                                                                        \
      __syncthreads();                                                         \
    } else {                                                                   \
      __syncthreads();                                                         \
      const int crow = tid >> 1, chf = tid & 1;                                \
      *(v16bf*)&btile[0][crow * 32 + chf * 16] =                               \
          *(const v16bf*)((Wpanel) + (size_t)crow * EMB + k0 + chf * 16);      \
      __syncthreads();                                                         \
    }                                                                          \
    const __bf16* bt = btile[cur];                                             \
    __builtin_prefetch(arow[0] + k0 + 256, 0, 1); /* global_prefetch_b8 */     \
    v16bf a[2], b[4];                                                          \
    _Pragma("unroll") for (int mt = 0; mt < 2; ++mt)                           \
        a[mt] = load_a_frag(arow[mt] + k0 + ka);                               \
    _Pragma("unroll") for (int nt = 0; nt < 4; ++nt)                           \
        b[nt] = load_b_frag(&bt[(wn * 64 + nt * 16 + l16) * 32 + kb]);         \
    _Pragma("unroll") for (int mt = 0; mt < 2; ++mt)                           \
        _Pragma("unroll") for (int nt = 0; nt < 4; ++nt)                       \
            acc[mt][nt] = wmma_bf16(a[mt], b[nt], acc[mt][nt]);                \
    if (HAVE_TDM) {                                                            \
      __syncthreads();                                                         \
      cur ^= 1;                                                                \
    }                                                                          \
  }

// ---------------------------------------------------------------------------
// QKV GEMM: C(8192x3072) = X @ Wqkv^T, scatter into Q(B,H,S,HD), K(B,H,S,HD),
// Vt(B,H,HD,S) (all bf16).
// ---------------------------------------------------------------------------
__global__ __launch_bounds__(256) void qkv_gemm(
    const __bf16* __restrict__ X, const __bf16* __restrict__ W,
    __bf16* __restrict__ Q, __bf16* __restrict__ K, __bf16* __restrict__ Vt) {
  const __bf16* wpanel = W + (size_t)(blockIdx.y * 128) * EMB;
  GEMM_MAINLOOP(X, wpanel, NQKV)

  // Scatter epilogue: a 16-wide, 16-aligned N tile lies entirely inside one
  // q/k/v segment (boundaries at multiples of 64 inside 192-chunks).
#pragma unroll
  for (int mt = 0; mt < 2; ++mt) {
#pragma unroll
    for (int nt = 0; nt < 4; ++nt) {
      const int n0  = Nbase + nt * 16;       // uniform
      const int h   = n0 / 192;
      const int t0  = n0 - h * 192;
      const int seg = t0 / 64;               // 0=q 1=k 2=v (uniform)
      const int d   = (t0 - seg * 64) + l16;
#pragma unroll
      for (int r = 0; r < 8; ++r) {
        const int m  = Mbase + mt * 16 + r + (hi ? 8 : 0);
        const int bb = m >> 11;              // /SEQ
        const int s  = m & (SEQ - 1);
        const __bf16 v = (__bf16)acc[mt][nt][r];
        if (seg == 0)
          Q[((size_t)(bb * NH + h) * SEQ + s) * HD + d] = v;
        else if (seg == 1)
          K[((size_t)(bb * NH + h) * SEQ + s) * HD + d] = v;
        else
          Vt[((size_t)(bb * NH + h) * HD + d) * SEQ + s] = v;
      }
    }
  }
}

// ---------------------------------------------------------------------------
// Projection GEMM: out(8192x1024 f32) = O @ Wp^T + bias
// ---------------------------------------------------------------------------
__global__ __launch_bounds__(256) void proj_gemm(
    const __bf16* __restrict__ A, const __bf16* __restrict__ W,
    const float* __restrict__ bias, float* __restrict__ out) {
  const __bf16* wpanel = W + (size_t)(blockIdx.y * 128) * EMB;
  GEMM_MAINLOOP(A, wpanel, EMB)

#pragma unroll
  for (int nt = 0; nt < 4; ++nt) {
    const int n = Nbase + nt * 16 + l16;
    const float bv = bias[n];
#pragma unroll
    for (int mt = 0; mt < 2; ++mt) {
#pragma unroll
      for (int r = 0; r < 8; ++r) {
        const int m = Mbase + mt * 16 + r + (hi ? 8 : 0);
        out[(size_t)m * EMB + n] = acc[mt][nt][r] + bv;
      }
    }
  }
}

// ---------------------------------------------------------------------------
// Flash attention: one wave per (b, h, 16-query tile); stream 32 keys/iter.
// Q,K: (B,H,S,HD) bf16; Vt: (B,H,HD,S) bf16; O: (B,S,E) bf16.
// ---------------------------------------------------------------------------
__global__ __launch_bounds__(256) void attn_kernel(
    const __bf16* __restrict__ Q, const __bf16* __restrict__ K,
    const __bf16* __restrict__ Vt, __bf16* __restrict__ O) {
  __shared__ __align__(64) __bf16 pbuf[8][16 * 32];  // per-wave P tile

  const int lane = threadIdx.x & 31;
  const int wave = threadIdx.x >> 5;
  const int l16  = lane & 15;
  const bool hi  = lane >= 16;

  const int id = blockIdx.x * 8 + wave;   // 0 .. B*H*(S/16)-1 = 8191
  const int qt = id & (SEQ / 16 - 1);
  const int h  = (id >> 7) & (NH - 1);
  const int b  = id >> 11;
  const int qbase = qt * 16;

  const __bf16* Qp  = Q + ((size_t)(b * NH + h) * SEQ) * HD;
  const __bf16* Kp  = K + ((size_t)(b * NH + h) * SEQ) * HD;
  const __bf16* Vtp = Vt + ((size_t)(b * NH + h) * HD) * SEQ;
  __bf16* pb = pbuf[wave];

  const __bf16* qrow = Qp + (size_t)(qbase + l16) * HD + (hi ? 16 : 0);
  const v16bf qf0 = load_a_frag(qrow);
  const v16bf qf1 = load_a_frag(qrow + 32);

  v8f o[4];
#pragma unroll
  for (int dt = 0; dt < 4; ++dt)
    o[dt] = (v8f){0.f, 0.f, 0.f, 0.f, 0.f, 0.f, 0.f, 0.f};
  float mrow[8], lrow[8];
#pragma unroll
  for (int r = 0; r < 8; ++r) { mrow[r] = -__builtin_inff(); lrow[r] = 0.f; }

  const float scale = 0.125f;      // HD^-0.5
  const int nkb = (qt + 2) >> 1;   // ceil((qt+1)/2) 32-key blocks

  for (int kbk = 0; kbk < nkb; ++kbk) {
    const int key0 = kbk * 32;
    v8f s0 = (v8f){0.f, 0.f, 0.f, 0.f, 0.f, 0.f, 0.f, 0.f};
    v8f s1 = s0;
    {
      const __bf16* kr0 = Kp + (size_t)(key0 + l16) * HD + (hi ? 16 : 0);
      const __bf16* kr1 = Kp + (size_t)(key0 + 16 + l16) * HD + (hi ? 16 : 0);
      s0 = wmma_bf16(qf0, load_b_frag(kr0), s0);
      s0 = wmma_bf16(qf1, load_b_frag(kr0 + 32), s0);
      s1 = wmma_bf16(qf0, load_b_frag(kr1), s1);
      s1 = wmma_bf16(qf1, load_b_frag(kr1 + 32), s1);
    }
#pragma unroll
    for (int r = 0; r < 8; ++r) {
      const int qg  = qbase + r + (hi ? 8 : 0);
      const int kg0 = key0 + l16;
      float x0 = s0[r] * scale;
      float x1 = s1[r] * scale;
      if (kg0 > qg)      x0 = -__builtin_inff();
      if (kg0 + 16 > qg) x1 = -__builtin_inff();
      float t = fmaxf(x0, x1);
      t = fmaxf(t, __shfl_xor(t, 1));
      t = fmaxf(t, __shfl_xor(t, 2));
      t = fmaxf(t, __shfl_xor(t, 4));
      t = fmaxf(t, __shfl_xor(t, 8));
      const float mn = fmaxf(mrow[r], t);
      const float al = __expf(mrow[r] - mn);
      const float p0 = __expf(x0 - mn);
      const float p1 = __expf(x1 - mn);
      float rs = p0 + p1;
      rs += __shfl_xor(rs, 1);
      rs += __shfl_xor(rs, 2);
      rs += __shfl_xor(rs, 4);
      rs += __shfl_xor(rs, 8);
      lrow[r] = lrow[r] * al + rs;
      mrow[r] = mn;
#pragma unroll
      for (int dt = 0; dt < 4; ++dt) o[dt][r] *= al;
      const int row = r + (hi ? 8 : 0);
      pb[row * 32 + l16]      = (__bf16)p0;
      pb[row * 32 + l16 + 16] = (__bf16)p1;
    }
    asm volatile("" ::: "memory");  // keep DS store->load ordering
    const v16bf pf = load_a_frag(pb + l16 * 32 + (hi ? 16 : 0));
#pragma unroll
    for (int dt = 0; dt < 4; ++dt) {
      const __bf16* vp =
          Vtp + (size_t)(dt * 16 + l16) * SEQ + key0 + (hi ? 16 : 0);
      o[dt] = wmma_bf16(pf, load_b_frag(vp), o[dt]);
    }
  }

#pragma unroll
  for (int r = 0; r < 8; ++r) {
    const int row = r + (hi ? 8 : 0);
    const float inv = 1.0f / lrow[r];
    const size_t base = ((size_t)b * SEQ + qbase + row) * EMB + h * HD + l16;
#pragma unroll
    for (int dt = 0; dt < 4; ++dt)
      O[base + dt * 16] = (__bf16)(o[dt][r] * inv);
  }
}

// ---------------------------------------------------------------------------
extern "C" void kernel_launch(void* const* d_in, const int* in_sizes, int n_in,
                              void* d_out, int out_size, void* d_ws,
                              size_t ws_size, hipStream_t stream) {
  (void)in_sizes; (void)n_in; (void)out_size; (void)ws_size;
  const float* x      = (const float*)d_in[0];
  const float* w_qkv  = (const float*)d_in[1];
  const float* w_proj = (const float*)d_in[2];
  const float* b_proj = (const float*)d_in[3];
  float* out = (float*)d_out;

  // workspace partition (all offsets 32B-aligned; total 88 MiB)
  char* ws = (char*)d_ws;
  const size_t nX  = (size_t)ROWS * EMB;             // 8388608
  const size_t nWq = (size_t)NQKV * EMB;             // 3145728
  const size_t nWp = (size_t)EMB * EMB;              // 1048576
  const size_t nH  = (size_t)BATCH * NH * SEQ * HD;  // 8388608
  __bf16* xb  = (__bf16*)(ws);
  __bf16* wqb = (__bf16*)(ws + 2 * nX);
  __bf16* wpb = (__bf16*)(ws + 2 * (nX + nWq));
  __bf16* Qb  = (__bf16*)(ws + 2 * (nX + nWq + nWp));
  __bf16* Kb  = (__bf16*)(ws + 2 * (nX + nWq + nWp + nH));
  __bf16* Vtb = (__bf16*)(ws + 2 * (nX + nWq + nWp + 2 * nH));
  __bf16* Ob  = (__bf16*)(ws + 2 * (nX + nWq + nWp + 3 * nH));

  cvt_f32_bf16<<<(int)(nX / 256), 256, 0, stream>>>(x, xb, (int)nX);
  cvt_f32_bf16<<<(int)(nWq / 256), 256, 0, stream>>>(w_qkv, wqb, (int)nWq);
  cvt_f32_bf16<<<(int)(nWp / 256), 256, 0, stream>>>(w_proj, wpb, (int)nWp);

  qkv_gemm<<<dim3(ROWS / 128, NQKV / 128), 256, 0, stream>>>(xb, wqb, Qb, Kb,
                                                             Vtb);
  attn_kernel<<<(BATCH * NH * (SEQ / 16)) / 8, 256, 0, stream>>>(Qb, Kb, Vtb,
                                                                 Ob);
  proj_gemm<<<dim3(ROWS / 128, EMB / 128), 256, 0, stream>>>(Ob, wpb, b_proj,
                                                             out);
}
